// AbsDiagNet_10462540333632
// MI455X (gfx1250) — matrix operand order repaired and checked by reading, not compiled, not run on hardware
//
#include <hip/hip_runtime.h>

// Problem dims (fixed by the reference).
#define S_DIM 512
#define B_DIM 128
#define I_DIM 1024
#define H_DIM 2048
#define O_DIM 1024
#define SC    64          // S-chunk: per-chunk xW (64*128*2048*4 = 67 MB) stays in 192 MB L2

typedef __attribute__((ext_vector_type(16))) __bf16 v16bf;
typedef __attribute__((ext_vector_type(8)))  float  v8f;

__device__ __forceinline__ unsigned short f32_to_bf16_rne(float f) {
  unsigned int u = __float_as_uint(f);
  u += 0x7FFFu + ((u >> 16) & 1u);   // round-to-nearest-even
  return (unsigned short)(u >> 16);
}
__device__ __forceinline__ unsigned int pack_bf16x2(float a, float b) {
  return (unsigned int)f32_to_bf16_rne(a) | ((unsigned int)f32_to_bf16_rne(b) << 16);
}

// ---------------------------------------------------------------------------
// C[M,N] = Abf[M,K] * Wbf[N,K]^T (+ optional bias); both operands bf16,
// f32 accumulate via v_wmma_f32_16x16x32_bf16.
// Block: 256 threads (8 wave32). Tile: MT=128, NT=64, K-step 32.
// Wave w owns rows [w*16, w*16+16) and four 16x16 N-subtiles (4 wmmas/step).
// Software pipeline: global loads for tile k+1 staged in VGPRs while tile k
// is consumed from LDS by the WMMAs.
// ---------------------------------------------------------------------------
#define MT 128
#define NT 64
#define KT 32
#define LDS_STRIDE 40   // halfwords per row: 80B pitch -> 16B aligned, bank-friendly

__global__ void __launch_bounds__(256)
gemm_bf16_wmma(const unsigned short* __restrict__ Abf, int lda,
               const unsigned short* __restrict__ Wbf, int ldw,
               float* __restrict__ C, int ldc,
               int K, const float* __restrict__ bias) {
  __shared__ unsigned short ldsA[MT * LDS_STRIDE];
  __shared__ unsigned short ldsB[NT * LDS_STRIDE];

  const int tid   = threadIdx.x;
  const int lane  = tid & 31;
  const int wave  = tid >> 5;
  const int lm    = lane & 15;   // M index (A) or N index (B/C) within a 16-tile
  const int khalf = lane >> 4;   // which K/M half this lane covers (ISA 7.12.2)

  const int blockM = blockIdx.y * MT;
  const int blockN = blockIdx.x * NT;

  // cooperative staging coords (computed once)
  const int ar  = (tid * 16) >> 5;   // A: 16 halfs/thread, MT*KT = 4096 halfs
  const int ac0 = (tid * 16) & 31;
  const int br  = (tid * 8) >> 5;    // B: 8 halfs/thread,  NT*KT = 2048 halfs
  const int bc0 = (tid * 8) & 31;

  uint4 aR0, aR1, bR;                // register-staged tiles

  auto ld_tiles = [&](int kk) {
    const unsigned short* sa = Abf + (size_t)(blockM + ar) * lda + kk + ac0;
    aR0 = *(const uint4*)sa;
    aR1 = *(const uint4*)(sa + 8);
    const unsigned short* sb = Wbf + (size_t)(blockN + br) * ldw + kk + bc0;
    bR  = *(const uint4*)sb;
  };
  auto st_tiles = [&]() {
    *(uint4*)&ldsA[ar * LDS_STRIDE + ac0]     = aR0;
    *(uint4*)&ldsA[ar * LDS_STRIDE + ac0 + 8] = aR1;
    *(uint4*)&ldsB[br * LDS_STRIDE + bc0]     = bR;
  };

  v8f acc[4] = {};

  ld_tiles(0);
  for (int kk = 0; kk < K; kk += KT) {
    st_tiles();
    __syncthreads();

    const int kn = kk + KT;
    if (kn < K) ld_tiles(kn);        // overlap global latency with WMMA work

    // ---- gather fragments per ISA 7.12.2 layouts ----
    union Frag { v16bf v; unsigned int u[8]; };
    Frag afrag, bfrag[4];
    const int arow = wave * 16 + lm;
#pragma unroll
    for (int t = 0; t < 8; ++t) {
      // A 16x32: VGPR t holds a K pair; lanes 0-15 low K-half, 16-31 high
      int kb = (t < 4) ? (2 * t + 8 * khalf) : (16 + 2 * (t - 4) + 8 * khalf);
      afrag.u[t] = *(const unsigned int*)&ldsA[arow * LDS_STRIDE + kb];
    }
#pragma unroll
    for (int n = 0; n < 4; ++n) {
#pragma unroll
      for (int t = 0; t < 8; ++t) {
        // B 32x16: lane = column N; same K order as A (lanes 0-15 K=0..7,16..23; lanes 16-31 K=8..15,24..31)
        int kb = (t < 4) ? (2 * t + 8 * khalf) : (16 + 2 * (t - 4) + 8 * khalf);
        bfrag[n].u[t] = *(const unsigned int*)&ldsB[(n * 16 + lm) * LDS_STRIDE + kb];
      }
    }
#pragma unroll
    for (int n = 0; n < 4; ++n) {
      acc[n] = __builtin_amdgcn_wmma_f32_16x16x32_bf16(false, afrag.v, false, bfrag[n].v,
                                                       (short)0, acc[n], false, false);
    }
    __syncthreads();
  }

  // ---- store C (D layout: lane = N col; VGPR r -> M = r + 8*khalf) ----
#pragma unroll
  for (int n = 0; n < 4; ++n) {
    const int ncol = blockN + n * 16 + lm;
    const float b = bias ? bias[ncol] : 0.0f;
#pragma unroll
    for (int r = 0; r < 8; ++r) {
      const int mrow = blockM + wave * 16 + r + 8 * khalf;
      C[(size_t)mrow * ldc + ncol] = acc[n][r] + b;
    }
  }
}

// ---------------------------------------------------------------------------
// h = |xW_t + HH*h| for t = 0..SC-1; one thread per (b,h), serial only in t.
// xW chunk is L2-resident (written by the GEMM just before on the same stream).
// ---------------------------------------------------------------------------
__global__ void __launch_bounds__(256)
recur_kernel(const float* __restrict__ xw, float* __restrict__ h,
             const float* __restrict__ HH) {
  const int idx = blockIdx.x * blockDim.x + threadIdx.x;   // < B*H
  const size_t stride = (size_t)B_DIM * H_DIM;
  float hv = h[idx];
  const float hh = HH[idx & (H_DIM - 1)];
#pragma unroll 4
  for (int t = 0; t < SC; ++t) {
    int tp = (t + 8 < SC) ? (t + 8) : (SC - 1);
    __builtin_prefetch(xw + (size_t)tp * stride + idx, 0, 0);  // global_prefetch_b8
    hv = fabsf(xw[(size_t)t * stride + idx] + hh * hv);
  }
  h[idx] = hv;
}

__global__ void __launch_bounds__(256)
cvt_f32_bf16(const float* __restrict__ src, unsigned short* __restrict__ dst, int n4) {
  const int i = blockIdx.x * blockDim.x + threadIdx.x;
  if (i < n4) {
    float4 f = ((const float4*)src)[i];
    uint2 p;
    p.x = pack_bf16x2(f.x, f.y);
    p.y = pack_bf16x2(f.z, f.w);
    ((uint2*)dst)[i] = p;
  }
}

__global__ void __launch_bounds__(256)
zero_f32(float* __restrict__ p, int n4) {
  const int i = blockIdx.x * blockDim.x + threadIdx.x;
  if (i < n4) {
    float4 z = {0.0f, 0.0f, 0.0f, 0.0f};
    ((float4*)p)[i] = z;
  }
}

// ---------------------------------------------------------------------------
extern "C" void kernel_launch(void* const* d_in, const int* in_sizes, int n_in,
                              void* d_out, int out_size, void* d_ws, size_t ws_size,
                              hipStream_t stream) {
  (void)in_sizes; (void)n_in; (void)out_size; (void)ws_size;

  const float* X    = (const float*)d_in[0];   // [S,B,I]
  const float* W_ih = (const float*)d_in[1];   // [H,I]
  const float* HH   = (const float*)d_in[2];   // [H]
  const float* W_ho = (const float*)d_in[3];   // [O,H]
  const float* b_ho = (const float*)d_in[4];   // [O]
  float* out = (float*)d_out;                  // [B,O] f32

  // Workspace: | xw chunk f32 | X chunk bf16 | h f32 | h bf16 | Wih bf16 | Who bf16 |
  char* ws = (char*)d_ws;
  float*          xw_buf  = (float*)ws;
  unsigned short* xbf     = (unsigned short*)(ws + (size_t)SC * B_DIM * H_DIM * sizeof(float));
  float*          h_state = (float*)((char*)xbf + (size_t)SC * B_DIM * I_DIM * sizeof(unsigned short));
  unsigned short* h_bf    = (unsigned short*)((char*)h_state + (size_t)B_DIM * H_DIM * sizeof(float));
  unsigned short* wih_bf  = h_bf + (size_t)B_DIM * H_DIM;
  unsigned short* who_bf  = wih_bf + (size_t)H_DIM * I_DIM;

  {
    int n4 = (H_DIM * I_DIM) / 4;
    cvt_f32_bf16<<<(n4 + 255) / 256, 256, 0, stream>>>(W_ih, wih_bf, n4);
  }
  {
    int n4 = (O_DIM * H_DIM) / 4;
    cvt_f32_bf16<<<(n4 + 255) / 256, 256, 0, stream>>>(W_ho, who_bf, n4);
  }
  {
    int n4 = (B_DIM * H_DIM) / 4;
    zero_f32<<<(n4 + 255) / 256, 256, 0, stream>>>(h_state, n4);
  }

  const dim3 block(256);
  for (int c = 0; c < S_DIM / SC; ++c) {
    // 1) convert X chunk to bf16 once (removes per-N-block redundant conversion)
    {
      int n4 = (SC * B_DIM * I_DIM) / 4;
      cvt_f32_bf16<<<(n4 + 255) / 256, 256, 0, stream>>>(
          X + (size_t)c * SC * B_DIM * I_DIM, xbf, n4);
    }
    // 2) GEMM: M = SC*B = 8192 contiguous (s,b) rows, N = H, K = I
    {
      dim3 grid(H_DIM / NT, (SC * B_DIM) / MT);          // 32 x 64 blocks
      gemm_bf16_wmma<<<grid, block, 0, stream>>>(xbf, I_DIM, wih_bf, I_DIM,
                                                 xw_buf, H_DIM, I_DIM, nullptr);
    }
    // 3) advance the recurrence SC steps (xw chunk still in L2)
    recur_kernel<<<(B_DIM * H_DIM) / 256, 256, 0, stream>>>(xw_buf, h_state, HH);
  }

  {
    // out[B,O] = h[B,H] @ W_ho^T + b_ho : M = 128, N = 1024, K = 2048
    int n4 = (B_DIM * H_DIM) / 4;
    cvt_f32_bf16<<<(n4 + 255) / 256, 256, 0, stream>>>(h_state, h_bf, n4);
    dim3 grid(O_DIM / NT, B_DIM / MT);                   // 16 x 1 blocks
    gemm_bf16_wmma<<<grid, block, 0, stream>>>(h_bf, H_DIM, who_bf, H_DIM,
                                               out, O_DIM, H_DIM, b_ho);
  }
}
